// DaleDendriticMLP_16183436771339
// MI455X (gfx1250) — matrix operand order repaired and checked by reading, not compiled, not run on hardware
//
#include <hip/hip_runtime.h>
#include <hip/hip_bf16.h>

typedef __attribute__((ext_vector_type(16))) __bf16 v16bf;
typedef __attribute__((ext_vector_type(8)))  float  v8f;

#define BB   256
#define DIN  2048
#define HID  2048
#define NSEG 10
#define DCTX 1024
#define OUTN 100
#define KSEL 102

// ---------------------------------------------------------------------------
// Fragment loaders: fp32 global memory -> bf16 WMMA fragments (mask fused).
// ---------------------------------------------------------------------------
__device__ inline void load8f(const float* __restrict__ p, float* t) {
  const float4* q = (const float4*)p;
  float4 f0 = q[0], f1 = q[1];
  t[0]=f0.x; t[1]=f0.y; t[2]=f0.z; t[3]=f0.w;
  t[4]=f1.x; t[5]=f1.y; t[6]=f1.z; t[7]=f1.w;
}

// A fragment, 16x32 bf16 (ISA 7.12.2): lane = (m&15) + 16*hi ;
// elem j -> K = hi*8 + j (j<8), K = 16 + hi*8 + (j-8) (j>=8).
__device__ inline v16bf load_afrag(const float* __restrict__ row, int k0, int hi) {
  float t[16];
  load8f(row + k0 + hi * 8,      t);
  load8f(row + k0 + 16 + hi * 8, t + 8);
  v16bf a;
#pragma unroll
  for (int i = 0; i < 16; ++i) a[i] = (__bf16)t[i];
  return a;
}

// B fragment, 32x16 bf16: lane = (n&15) + 16*hi ; elem j -> K = hi*8 + j (j<8), K = 16 + hi*8 + (j-8) (j>=8).
// Mask multiply fused here (v_pk_mul_f32 + v_cvt_pk_bf16_f32).
__device__ inline v16bf load_bfrag(const float* __restrict__ row,
                                   const float* __restrict__ mrow,
                                   int k0, int hi) {
  float t[16];
  const int off = k0 + hi * 8;
  load8f(row + off,      t);
  load8f(row + off + 16, t + 8);
  if (mrow) {
    float m[16];
    load8f(mrow + off,      m);
    load8f(mrow + off + 16, m + 8);
#pragma unroll
    for (int i = 0; i < 16; ++i) t[i] *= m[i];
  }
  v16bf b;
#pragma unroll
  for (int i = 0; i < 16; ++i) b[i] = (__bf16)t[i];
  return b;
}

// ---------------------------------------------------------------------------
// Y[M,N] = A[M,K] @ (W[N,K] * Mask[N,K])^T + bias
// Block: 8 waves, tile 64 x (32*NT).  Each wave: one 16-row x (16*NT)-col
// strip -> NT accumulators sharing a single A fragment per K step
// (v_wmma_f32_16x16x32_bf16).  Grid: (N/(32*NT), M/64), 256 threads.
// ---------------------------------------------------------------------------
template <int NT>
__global__ __launch_bounds__(256)
void gemm_bf16_wmma(float* __restrict__ Y,
                    const float* __restrict__ A,
                    const float* __restrict__ W,
                    const float* __restrict__ Mask,
                    const float* __restrict__ bias,
                    int M, int N, int K) {
  const int wave = threadIdx.x >> 5;
  const int lane = threadIdx.x & 31;
  const int lm   = lane & 15;
  const int hi   = lane >> 4;

  const int mi = wave & 3;   // 0..3 : m-tile inside block
  const int nh = wave >> 2;  // 0..1 : column-half inside block

  const int m0 = blockIdx.y * 64 + mi * 16;
  const int n0 = blockIdx.x * (32 * NT) + nh * (16 * NT);

  const float* Arow = A + (size_t)(m0 + lm) * K;
  const float* Wrow[NT];
  const float* Mrow[NT];
#pragma unroll
  for (int t = 0; t < NT; ++t) {
    Wrow[t] = W + (size_t)(n0 + 16 * t + lm) * K;
    Mrow[t] = Mask ? Mask + (size_t)(n0 + 16 * t + lm) * K : nullptr;
  }

  v8f acc[NT];
#pragma unroll
  for (int t = 0; t < NT; ++t) acc[t] = (v8f){};

  for (int k0 = 0; k0 < K; k0 += 32) {
    if (k0 + 32 < K) {  // stream hint on the HBM-bound weight side
#pragma unroll
      for (int t = 0; t < NT; ++t) __builtin_prefetch(Wrow[t] + k0 + 32, 0, 1);
    }
    const v16bf a = load_afrag(Arow, k0, hi);
#pragma unroll
    for (int t = 0; t < NT; ++t) {
      v16bf b = load_bfrag(Wrow[t], Mrow[t], k0, hi);
      acc[t] = __builtin_amdgcn_wmma_f32_16x16x32_bf16(false, a, false, b,
                                                       (short)0, acc[t],
                                                       false, false);
    }
  }

  // C/D layout: lane half 'hi' holds rows m0 + hi*8 + r, column = lm.
#pragma unroll
  for (int t = 0; t < NT; ++t) {
    const int nc = n0 + 16 * t + lm;
    const float bv = bias ? bias[nc] : 0.0f;
#pragma unroll
    for (int r = 0; r < 8; ++r) {
      const int row = m0 + hi * 8 + r;
      Y[(size_t)row * N + nc] = acc[t][r] + bv;
    }
  }
}

// ---------------------------------------------------------------------------
// Dendritic gating: for each (b,u) pick segment with max |d|, y *= sigmoid(d).
// d laid out [b][u*NSEG + s] (GEMM N index = u*NSEG+s).
// ---------------------------------------------------------------------------
__global__ __launch_bounds__(256)
void gate_kernel(float* __restrict__ y, const float* __restrict__ d, int BH) {
  const int i = blockIdx.x * blockDim.x + threadIdx.x;
  if (i >= BH) return;
  const float* dp = d + (size_t)i * NSEG;
  float best  = dp[0];
  float besta = fabsf(best);
#pragma unroll
  for (int s = 1; s < NSEG; ++s) {
    float v = dp[s], a = fabsf(v);
    if (a > besta) { besta = a; best = v; }
  }
  y[i] *= 1.0f / (1.0f + expf(-best));
}

// ---------------------------------------------------------------------------
// k-winners: per row, keep top-KSEL of HID values, zero the rest.
// Radix select (4x8-bit passes) on sign-flipped float keys; deterministic.
// One block of 256 threads per row; each thread owns 8 elements.
// ---------------------------------------------------------------------------
__global__ __launch_bounds__(256)
void kwinners_kernel(float* __restrict__ h, const float* __restrict__ y) {
  __shared__ int      hist[256];
  __shared__ unsigned s_prefix;
  __shared__ int      s_remain;

  const int b = blockIdx.x;
  const float* yr = y + (size_t)b * HID;
  float*       hr = h + (size_t)b * HID;

  unsigned keys[8];
  float    vals[8];
#pragma unroll
  for (int j = 0; j < 8; ++j) {
    float v = yr[threadIdx.x + j * 256];
    vals[j] = v;
    unsigned u = __float_as_uint(v);
    keys[j] = (u & 0x80000000u) ? ~u : (u | 0x80000000u);  // monotone key
  }
  if (threadIdx.x == 0) { s_prefix = 0u; s_remain = KSEL; }
  __syncthreads();

  for (int pass = 0; pass < 4; ++pass) {
    const int shift = 24 - pass * 8;
    hist[threadIdx.x] = 0;
    __syncthreads();
    const unsigned prefix = s_prefix;
    const unsigned hm = (pass == 0) ? 0u : (0xFFFFFFFFu << (shift + 8));
#pragma unroll
    for (int j = 0; j < 8; ++j)
      if ((keys[j] & hm) == prefix)
        atomicAdd(&hist[(keys[j] >> shift) & 255], 1);
    __syncthreads();
    if (threadIdx.x == 0) {
      int rem = s_remain, dsel = 0;
      for (int dgt = 255; dgt >= 0; --dgt) {
        int c = hist[dgt];
        if (c >= rem) { dsel = dgt; break; }
        rem -= c;
      }
      s_remain = rem;
      s_prefix = prefix | ((unsigned)dsel << shift);
    }
    __syncthreads();
  }

  const unsigned thr = s_prefix;  // key of the KSEL-th largest value
#pragma unroll
  for (int j = 0; j < 8; ++j)
    hr[threadIdx.x + j * 256] = (keys[j] >= thr) ? vals[j] : 0.0f;
}

// ---------------------------------------------------------------------------
// Dale head: W_out = Wex - Wei @ Wix  (rank-1), then out = h @ W_out^T + b.
// ---------------------------------------------------------------------------
__global__ __launch_bounds__(256)
void wout_kernel(float* __restrict__ Wout, const float* __restrict__ Wex,
                 const float* __restrict__ Wix, const float* __restrict__ Wei) {
  const int i = blockIdx.x * blockDim.x + threadIdx.x;
  if (i >= OUTN * HID) return;
  const int o = i / HID, c = i % HID;
  Wout[i] = Wex[i] - Wei[o] * Wix[c];
}

__global__ __launch_bounds__(256)
void head_kernel(float* __restrict__ out, const float* __restrict__ h,
                 const float* __restrict__ Wout, const float* __restrict__ bout) {
  const int i = blockIdx.x * blockDim.x + threadIdx.x;
  if (i >= BB * OUTN) return;
  const int b = i / OUTN, o = i % OUTN;
  const float4* hp = (const float4*)(h    + (size_t)b * HID);
  const float4* wp = (const float4*)(Wout + (size_t)o * HID);
  float acc = 0.0f;
  for (int k = 0; k < HID / 4; ++k) {
    float4 a = hp[k], w = wp[k];
    acc += a.x * w.x + a.y * w.y + a.z * w.z + a.w * w.w;
  }
  out[i] = acc + bout[o];
}

// ---------------------------------------------------------------------------
extern "C" void kernel_launch(void* const* d_in, const int* in_sizes, int n_in,
                              void* d_out, int out_size, void* d_ws, size_t ws_size,
                              hipStream_t stream) {
  const float* x     = (const float*)d_in[0];
  const float* ctx   = (const float*)d_in[1];
  const float* W1    = (const float*)d_in[2];
  const float* b1    = (const float*)d_in[3];
  const float* segW1 = (const float*)d_in[4];
  const float* mW1   = (const float*)d_in[5];
  const float* mS1   = (const float*)d_in[6];
  const float* W2    = (const float*)d_in[7];
  const float* b2    = (const float*)d_in[8];
  const float* segW2 = (const float*)d_in[9];
  const float* mW2   = (const float*)d_in[10];
  const float* mS2   = (const float*)d_in[11];
  const float* Wex   = (const float*)d_in[12];
  const float* Wix   = (const float*)d_in[13];
  const float* Wei   = (const float*)d_in[14];
  const float* bo    = (const float*)d_in[15];
  float* out = (float*)d_out;

  float* ws   = (float*)d_ws;
  float* y    = ws;                         // [256, 2048]
  float* dbuf = y    + (size_t)BB * HID;    // [256, 20480]
  float* h1   = dbuf + (size_t)BB * HID * NSEG;
  float* h2   = h1   + (size_t)BB * HID;
  float* Wout = h2   + (size_t)BB * HID;    // [100, 2048]

  const dim3 blk(256);
  const dim3 gFF(HID / 64, BB / 64);               // NT=2 -> (32, 4)
  const dim3 gDN(HID * NSEG / 128, BB / 64);       // NT=4 -> (160, 4)

  // ---- layer 1 ----
  gemm_bf16_wmma<2><<<gFF, blk, 0, stream>>>(y,    x,   W1,    mW1, b1,      BB, HID,        DIN);
  gemm_bf16_wmma<4><<<gDN, blk, 0, stream>>>(dbuf, ctx, segW1, mS1, nullptr, BB, HID * NSEG, DCTX);
  gate_kernel    <<<BB * HID / 256, blk, 0, stream>>>(y, dbuf, BB * HID);
  kwinners_kernel<<<BB, blk, 0, stream>>>(h1, y);

  // ---- layer 2 ----
  gemm_bf16_wmma<2><<<gFF, blk, 0, stream>>>(y,    h1,  W2,    mW2, b2,      BB, HID,        HID);
  gemm_bf16_wmma<4><<<gDN, blk, 0, stream>>>(dbuf, ctx, segW2, mS2, nullptr, BB, HID * NSEG, DCTX);
  gate_kernel    <<<BB * HID / 256, blk, 0, stream>>>(y, dbuf, BB * HID);
  kwinners_kernel<<<BB, blk, 0, stream>>>(h2, y);

  // ---- Dale output head ----
  wout_kernel<<<(OUTN * HID + 255) / 256, blk, 0, stream>>>(Wout, Wex, Wix, Wei);
  head_kernel<<<(BB * OUTN + 255) / 256, blk, 0, stream>>>(out, h2, Wout, bo);
}
